// BidirectionalMambaBlock_1219770712822
// MI455X (gfx1250) — matrix operand order repaired and checked by reading, not compiled, not run on hardware
//
#include <hip/hip_runtime.h>
#include <cstdint>

// ---------------------------------------------------------------------------
// Bidirectional Mamba block for MI455X (gfx1250, wave32).
// GEMMs: v_wmma_f32_16x16x32_bf16, fp32 tiles staged into LDS with
// global_load_async_to_lds_b128 (double-buffered, ASYNCcnt), operands
// converted fp32->bf16 at fragment build via packed hardware cvt.
// Conv / selective-scan / LayerNorm are VALU kernels.
// ---------------------------------------------------------------------------

typedef __attribute__((ext_vector_type(16))) __bf16 v16bf;
typedef __attribute__((ext_vector_type(2)))  __bf16 v2bf;
typedef __attribute__((ext_vector_type(8)))  float  v8f;

namespace cfg {
constexpr int cB = 16, cN = 128, cL = 512, cD = 1024, cS = 16, cK = 4, cR = 64, cH = 2048;
constexpr int BM = 128, BN = 128, BK = 32;
constexpr int LDF = BK + 4;   // fp32 row stride in LDS (pad 16B to stagger banks)
}
using namespace cfg;

__device__ __forceinline__ unsigned pkbf(float lo, float hi) {
  v2bf p = { (__bf16)lo, (__bf16)hi };      // -> v_cvt_pk_bf16_f32
  return __builtin_bit_cast(unsigned, p);
}

struct U8x4 { unsigned v[8]; };             // 32 bytes == v16bf

// Convert two 8-float runs (16B-aligned) into one v16bf fragment.
__device__ __forceinline__ v16bf frag_cvt(const float* p0, const float* p1) {
  float4 a0 = *(const float4*)p0;
  float4 a1 = *(const float4*)(p0 + 4);
  float4 b0 = *(const float4*)p1;
  float4 b1 = *(const float4*)(p1 + 4);
  U8x4 u;
  u.v[0] = pkbf(a0.x, a0.y); u.v[1] = pkbf(a0.z, a0.w);
  u.v[2] = pkbf(a1.x, a1.y); u.v[3] = pkbf(a1.z, a1.w);
  u.v[4] = pkbf(b0.x, b0.y); u.v[5] = pkbf(b0.z, b0.w);
  u.v[6] = pkbf(b1.x, b1.y); u.v[7] = pkbf(b1.z, b1.w);
  return __builtin_bit_cast(v16bf, u);
}

__device__ __forceinline__ void async_b128(unsigned ldsByteAddr, const float* gaddr) {
  asm volatile("global_load_async_to_lds_b128 %0, %1, off"
               :: "v"(ldsByteAddr), "v"((unsigned long long)(uintptr_t)gaddr)
               : "memory");
}
__device__ __forceinline__ void wait_async0() {
  asm volatile("s_wait_asynccnt 0" ::: "memory");
}

// epi: 0 = none, 1 = +bias, 2 = relu(x+bias), 3 = softplus(x+bias)
// A: (M,K) row-major stride lda; W: (Nn,K) row-major stride ldb (B = W^T);
// C: (M,Nn) row-major stride ldc.
__global__ void __launch_bounds__(256)
gemm_wmma_bf16(const float* __restrict__ A, const float* __restrict__ W,
               const float* __restrict__ bias, float* __restrict__ C,
               int M, int Nn, int Kk, int lda, int ldb, int ldc, int epi)
{
  __shared__ __align__(16) float Asf[2][BM * LDF];
  __shared__ __align__(16) float Bsf[2][BN * LDF];

  const int tid  = threadIdx.x;
  const int lane = tid & 31;
  const int wave = tid >> 5;
  const int wm   = wave & 3;      // 4 waves along M (32 rows each)
  const int wn   = wave >> 2;     // 2 waves along N (64 cols each)
  const int bm   = blockIdx.y * BM;
  const int bn   = blockIdx.x * BN;
  const int r15  = lane & 15;
  const int hsel = lane >> 4;

  v8f acc[2][4];
#pragma unroll
  for (int i = 0; i < 2; i++)
#pragma unroll
    for (int j = 0; j < 4; j++)
      acc[i][j] = (v8f){0.f,0.f,0.f,0.f,0.f,0.f,0.f,0.f};

  // Issue one tile's worth of async copies (4 A-chunks + 4 B-chunks / thread).
  auto stage = [&](int kt, int buf) {
#pragma unroll
    for (int i = 0; i < 4; i++) {
      int lin = i * 256 + tid;            // 1024 float4 chunks per tile
      int row = lin >> 3;
      int c4  = (lin & 7) * 4;
      // A tile: rows always valid (M % 128 == 0)
      async_b128((unsigned)(uintptr_t)&Asf[buf][row * LDF + c4],
                 A + (size_t)(bm + row) * lda + kt + c4);
      // B tile: clamp partial-N rows (outputs for those cols are not stored)
      int wr = bn + row; if (wr >= Nn) wr = Nn - 1;
      async_b128((unsigned)(uintptr_t)&Bsf[buf][row * LDF + c4],
                 W + (size_t)wr * ldb + kt + c4);
    }
  };

  const int NT = Kk / BK;
  stage(0, 0);

  for (int t = 0; t < NT; ++t) {
    const int buf = t & 1;
    wait_async0();
    __syncthreads();                       // tile t resident for all waves
    if (t + 1 < NT) stage((t + 1) * BK, buf ^ 1);

    const float* as = &Asf[buf][0];
    const float* bs = &Bsf[buf][0];

    // A (16x32 bf16): lanes 0-15 -> K 0..7,16..23 ; lanes 16-31 -> K 8..15,24..31
    v16bf afr[2], bfr[4];
#pragma unroll
    for (int fm = 0; fm < 2; fm++) {
      int row = wm * 32 + fm * 16 + r15;
      int k0  = hsel * 8;
      afr[fm] = frag_cvt(as + row * LDF + k0, as + row * LDF + k0 + 16);
    }
    // B (32x16 bf16): lanes 0-15 -> K 0..7,16..23 of col n ; lanes 16-31 -> K 8..15,24..31
#pragma unroll
    for (int fn = 0; fn < 4; fn++) {
      int col = wn * 64 + fn * 16 + r15;
      int kb  = hsel * 8;
      bfr[fn] = frag_cvt(bs + col * LDF + kb, bs + col * LDF + kb + 16);
    }
#pragma unroll
    for (int fm = 0; fm < 2; fm++)
#pragma unroll
      for (int fn = 0; fn < 4; fn++)
        acc[fm][fn] = __builtin_amdgcn_wmma_f32_16x16x32_bf16(
            false, afr[fm], false, bfr[fn], (short)0, acc[fm][fn], false, false);
  }

  // ---- epilogue + store (C layout: lane = col, row = v + 8*(lane>>4)) ----
#pragma unroll
  for (int fm = 0; fm < 2; fm++) {
#pragma unroll
    for (int fn = 0; fn < 4; fn++) {
      int col = bn + wn * 64 + fn * 16 + r15;
      if (col >= Nn) continue;
      float bv = (epi != 0 && bias) ? bias[col] : 0.f;
      int rbase = bm + wm * 32 + fm * 16 + hsel * 8;
#pragma unroll
      for (int v = 0; v < 8; v++) {
        float val = acc[fm][fn][v];
        if (epi == 1)      { val += bv; }
        else if (epi == 2) { val += bv; val = fmaxf(val, 0.f); }
        else if (epi == 3) { val += bv; val = (val > 20.f) ? val : log1pf(__expf(val)); }
        C[(size_t)(rbase + v) * ldc + col] = val;
      }
    }
  }
}

// ---------------------------------------------------------------------------
__global__ void flip_kernel(const float* __restrict__ x, float* __restrict__ xf)
{
  int idx = blockIdx.x * blockDim.x + threadIdx.x;
  if (idx >= cB * cN * cL) return;
  int i = idx % cL;
  int n = (idx / cL) % cN;
  int b = idx / (cL * cN);
  xf[((size_t)b * cN + (cN - 1 - n)) * cL + i] = x[idx];
}

// depthwise causal conv (K=4) + SiLU; reads xi = cols [0,D) of proj (B*N,2D)
__global__ void conv_silu_kernel(const float* __restrict__ proj,
                                 const float* __restrict__ cw,
                                 const float* __restrict__ cb,
                                 float* __restrict__ xc)
{
  int idx = blockIdx.x * blockDim.x + threadIdx.x;
  if (idx >= cB * cN * cD) return;
  int d = idx % cD;
  int n = (idx / cD) % cN;
  int b = idx / (cD * cN);
  float acc = cb[d];
#pragma unroll
  for (int k = 0; k < cK; k++) {
    int nn = n - (cK - 1) + k;
    if (nn >= 0)
      acc += proj[((size_t)b * cN + nn) * (2 * cD) + d] * cw[d * cK + k];
  }
  xc[idx] = acc / (1.f + __expf(-acc));   // silu
}

// selective-scan: one thread per (b,d); S=16 state in registers.
// Reads xc (conv output), overwrites it in place with the gated output.
__global__ void scan_kernel(const float* __restrict__ proj,   // (B*N,2D): res in cols [D,2D)
                            float* __restrict__ xc,           // (B*N,D) in: conv; out: y
                            const float* __restrict__ dbc,    // (B*N,96): dt|Bm|Cm
                            const float* __restrict__ delta,  // (B*N,D)
                            const float* __restrict__ A_log,  // (D,S)
                            const float* __restrict__ Dp)     // (D)
{
  int idx = blockIdx.x * blockDim.x + threadIdx.x;
  if (idx >= cB * cD) return;
  int b = idx / cD, d = idx % cD;
  float An[cS];
#pragma unroll
  for (int s = 0; s < cS; s++) An[s] = -__expf(A_log[d * cS + s]);
  float dpd = Dp[d];
  float h[cS];
#pragma unroll
  for (int s = 0; s < cS; s++) h[s] = 0.f;
  for (int n = 0; n < cN; n++) {
    size_t rn = (size_t)b * cN + n;
    float dl = delta[rn * cD + d];
    float xv = xc[rn * cD + d];
    const float* bm = dbc + rn * (cR + 2 * cS) + cR;
    const float* cm = bm + cS;
    float y = 0.f;
#pragma unroll
    for (int s = 0; s < cS; s++) {
      float dA = __expf(dl * An[s]);
      h[s] = dA * h[s] + dl * bm[s] * xv;
      y += h[s] * cm[s];
    }
    float res = proj[rn * (2 * cD) + cD + d];
    float sil = res / (1.f + __expf(-res));
    xc[rn * cD + d] = (y + xv * dpd) * sil;
  }
}

// out = LayerNorm(a + b2 [+ c (optionally seq-flipped)]) * g + beta
__global__ void __launch_bounds__(256)
add_ln_kernel(const float* __restrict__ a, const float* __restrict__ b2,
              const float* __restrict__ c, int revc,
              const float* __restrict__ g, const float* __restrict__ be,
              float* __restrict__ out)
{
  __shared__ float row[cL];
  __shared__ float red[256];
  int r = blockIdx.x;
  int bb = r / cN, n = r % cN;
  const float* cr = nullptr;
  if (c) cr = c + (size_t)(bb * cN + (revc ? (cN - 1 - n) : n)) * cL;
  const float* ar = a + (size_t)r * cL;
  const float* br = b2 + (size_t)r * cL;
  float lsum = 0.f;
  for (int i = threadIdx.x; i < cL; i += 256) {
    float v = ar[i] + br[i] + (cr ? cr[i] : 0.f);
    row[i] = v;
    lsum += v;
  }
  red[threadIdx.x] = lsum;
  __syncthreads();
  for (int s = 128; s > 0; s >>= 1) {
    if (threadIdx.x < s) red[threadIdx.x] += red[threadIdx.x + s];
    __syncthreads();
  }
  float mean = red[0] / cL;
  __syncthreads();
  float lsq = 0.f;
  for (int i = threadIdx.x; i < cL; i += 256) {
    float dv = row[i] - mean;
    lsq += dv * dv;
  }
  red[threadIdx.x] = lsq;
  __syncthreads();
  for (int s = 128; s > 0; s >>= 1) {
    if (threadIdx.x < s) red[threadIdx.x] += red[threadIdx.x + s];
    __syncthreads();
  }
  float inv = rsqrtf(red[0] / cL + 1e-5f);
  float* orow = out + (size_t)r * cL;
  for (int i = threadIdx.x; i < cL; i += 256)
    orow[i] = (row[i] - mean) * inv * g[i] + be[i];
}

// ---------------------------------------------------------------------------
extern "C" void kernel_launch(void* const* d_in, const int* in_sizes, int n_in,
                              void* d_out, int out_size, void* d_ws, size_t ws_size,
                              hipStream_t stream)
{
  (void)in_sizes; (void)n_in; (void)out_size; (void)ws_size;
  const float* x        = (const float*)d_in[0];
  const float* f_in_w   = (const float*)d_in[1];
  const float* f_conv_w = (const float*)d_in[2];
  const float* f_conv_b = (const float*)d_in[3];
  const float* f_xproj  = (const float*)d_in[4];
  const float* f_dt_w   = (const float*)d_in[5];
  const float* f_dt_b   = (const float*)d_in[6];
  const float* f_A_log  = (const float*)d_in[7];
  const float* f_Dp     = (const float*)d_in[8];
  const float* f_out_w  = (const float*)d_in[9];
  const float* r_in_w   = (const float*)d_in[10];
  const float* r_conv_w = (const float*)d_in[11];
  const float* r_conv_b = (const float*)d_in[12];
  const float* r_xproj  = (const float*)d_in[13];
  const float* r_dt_w   = (const float*)d_in[14];
  const float* r_dt_b   = (const float*)d_in[15];
  const float* r_A_log  = (const float*)d_in[16];
  const float* r_Dp     = (const float*)d_in[17];
  const float* r_out_w  = (const float*)d_in[18];
  const float* pu_w     = (const float*)d_in[19];
  const float* pu_b     = (const float*)d_in[20];
  const float* pl_w     = (const float*)d_in[21];
  const float* pl_b     = (const float*)d_in[22];
  const float* ln_g     = (const float*)d_in[23];
  const float* ln_b     = (const float*)d_in[24];

  const int M = cB * cN;               // 2048 rows
  float* ws = (float*)d_ws;
  size_t off = 0;
  auto alloc = [&](size_t nel) { float* p = ws + off; off += nel; return p; };
  float* xflip = alloc((size_t)cB * cN * cL);
  float* projF = alloc((size_t)cB * cN * 2 * cD);
  float* projR = alloc((size_t)cB * cN * 2 * cD);
  float* xcF   = alloc((size_t)cB * cN * cD);
  float* xcR   = alloc((size_t)cB * cN * cD);
  float* dbcF  = alloc((size_t)cB * cN * (cR + 2 * cS));
  float* dbcR  = alloc((size_t)cB * cN * (cR + 2 * cS));
  float* dltF  = alloc((size_t)cB * cN * cD);
  float* dltR  = alloc((size_t)cB * cN * cD);
  float* y1    = alloc((size_t)cB * cN * cL);
  float* y2    = alloc((size_t)cB * cN * cL);
  float* y3    = alloc((size_t)cB * cN * cL);
  float* hid   = projF;                // reuse: proj buffers dead after scan
  float* yp    = projR;

  dim3 blk(256);
  auto gemm = [&](const float* A, const float* W, const float* bias, float* C,
                  int Nn, int Kk, int lda, int ldb, int ldc, int epi) {
    dim3 grid((Nn + BN - 1) / BN, M / BM);
    gemm_wmma_bf16<<<grid, blk, 0, stream>>>(A, W, bias, C, M, Nn, Kk, lda, ldb, ldc, epi);
  };

  flip_kernel<<<(cB * cN * cL) / 256, blk, 0, stream>>>(x, xflip);

  // in_proj: (2048,512) @ (2048,512)^T -> (2048,2048) = [xi | res]
  gemm(x,     f_in_w, nullptr, projF, 2 * cD, cL, cL, cL, 2 * cD, 0);
  gemm(xflip, r_in_w, nullptr, projR, 2 * cD, cL, cL, cL, 2 * cD, 0);

  conv_silu_kernel<<<(cB * cN * cD) / 256, blk, 0, stream>>>(projF, f_conv_w, f_conv_b, xcF);
  conv_silu_kernel<<<(cB * cN * cD) / 256, blk, 0, stream>>>(projR, r_conv_w, r_conv_b, xcR);

  // xproj: (2048,1024) -> (2048,96) = [dt(64) | Bm(16) | Cm(16)]
  gemm(xcF, f_xproj, nullptr, dbcF, cR + 2 * cS, cD, cD, cD, cR + 2 * cS, 0);
  gemm(xcR, r_xproj, nullptr, dbcR, cR + 2 * cS, cD, cD, cD, cR + 2 * cS, 0);

  // delta = softplus(dt @ dt_w^T + dt_b): (2048,64) -> (2048,1024)
  gemm(dbcF, f_dt_w, f_dt_b, dltF, cD, cR, cR + 2 * cS, cR, cD, 3);
  gemm(dbcR, r_dt_w, r_dt_b, dltR, cD, cR, cR + 2 * cS, cR, cD, 3);

  scan_kernel<<<(cB * cD) / 256, blk, 0, stream>>>(projF, xcF, dbcF, dltF, f_A_log, f_Dp);
  scan_kernel<<<(cB * cD) / 256, blk, 0, stream>>>(projR, xcR, dbcR, dltR, r_A_log, r_Dp);

  // out_proj: (2048,1024) -> (2048,512)
  gemm(xcF, f_out_w, nullptr, y1, cL, cD, cD, cD, cL, 0);
  gemm(xcR, r_out_w, nullptr, y2, cL, cD, cD, cD, cL, 0);

  // y3 = LN(x + y1 + flip(y2))
  add_ln_kernel<<<M, blk, 0, stream>>>(x, y1, y2, 1, ln_g, ln_b, y3);

  // MLP: relu(y3 @ pu^T + pu_b) @ pl^T + pl_b
  gemm(y3,  pu_w, pu_b, hid, cH, cL, cL, cL, cH, 2);
  gemm(hid, pl_w, pl_b, yp,  cL, cH, cH, cH, cL, 1);

  // out = LN(yp + y3)
  add_ln_kernel<<<M, blk, 0, stream>>>(yp, y3, nullptr, 0, ln_g, ln_b, (float*)d_out);
}
